// CrossYearEpisodicMemory_29953101923011
// MI455X (gfx1250) — matrix operand order repaired and checked by reading, not compiled, not run on hardware
//
#include <hip/hip_runtime.h>
#include <hip/hip_bf16.h>

// ---------------- problem constants ----------------
#define B_   64
#define T_   96
#define N_   325
#define D_   128
#define H_   4
#define DK_  32
#define K_   8
#define M_   768
#define BN_  (B_ * N_)        // 20800 = 325*64
#define BND_ (BN_ * D_)       // 2662400
#define D4_  512
#define NT_  21               // ceil(N/16)

static __device__ __constant__ float INV_SCALE = 0.17677669529663687f; // 1/sqrt(32)
#define TWO_PI_F 6.28318530717958647692f
#define INV_SQRT128 0.08838834764831845f

typedef __attribute__((ext_vector_type(16))) __bf16 v16bf;
typedef __attribute__((ext_vector_type(8)))  float  v8f;
typedef __attribute__((ext_vector_type(4)))  unsigned int u32x4;
typedef __attribute__((ext_vector_type(4)))  int i32x4;

#if defined(__HIP_DEVICE_COMPILE__) && \
    __has_builtin(__builtin_amdgcn_global_load_async_to_lds_b128) && \
    __has_builtin(__builtin_amdgcn_s_wait_asynccnt)
#define ASYNC_CP 1
#else
#define ASYNC_CP 0
#endif

__device__ __forceinline__ __bf16 bf(float x) { return (__bf16)x; }

__device__ __forceinline__ v8f wmma_bf16(v16bf a, v16bf b, v8f c) {
  return __builtin_amdgcn_wmma_f32_16x16x32_bf16(false, a, false, b, (short)0, c,
                                                 false, false);
}

#if ASYNC_CP
typedef __attribute__((address_space(1))) i32x4 gi32x4;
typedef __attribute__((address_space(3))) i32x4 li32x4;
__device__ __forceinline__ void async_b128(const __bf16* g, __bf16* l) {
  __builtin_amdgcn_global_load_async_to_lds_b128((gi32x4*)g, (li32x4*)l, 0, 0);
}
#endif

// A fragment 16x32 bf16, row-major src (16B-aligned rows): 16 contiguous elems.
// M=lane&15, K {half*16..+15} with half = lane>>4.
__device__ __forceinline__ v16bf load_A16(const __bf16* src, int ld) {
  int lane = threadIdx.x & 31, hf = lane >> 4, m = lane & 15;
  const __bf16* p = src + m * ld + hf * 16;
  union { v16bf v; u32x4 q[2]; } u;
  u.q[0] = *(const u32x4*)(p);
  u.q[1] = *(const u32x4*)(p + 8);
  return u.v;
}

// B fragment 32x16 bf16 from a TRANSPOSED tile (srcT[n][k], leading dim ld, ld%8==0):
// lane n=lane&15 reads 16 contiguous K values starting at (lane>>4)*16.
__device__ __forceinline__ v16bf load_Bt16(const __bf16* srcT, int ld) {
  int lane = threadIdx.x & 31, khalf = lane >> 4, n = lane & 15;
  const __bf16* p = srcT + n * ld + khalf * 16;
  union { v16bf v; u32x4 q[2]; } u;
  u.q[0] = *(const u32x4*)(p);
  u.q[1] = *(const u32x4*)(p + 8);
  return u.v;
}

// C/D 16x16 f32: VGPR r -> row (r + 8*(lane>>4)), col (lane&15).
__device__ __forceinline__ void store_c_bias_lds(float* dst, int ld, v8f c,
                                                 const float* bias, int colbase) {
  int lane = threadIdx.x & 31, hf = lane >> 4, col = lane & 15;
  float bv = bias[colbase + col];
#pragma unroll
  for (int r = 0; r < 8; ++r) dst[(r + hf * 8) * ld + col] = c[r] + bv;
}

__device__ __forceinline__ float wave_sum(float v) {
#pragma unroll
  for (int o = 16; o > 0; o >>= 1) v += __shfl_xor(v, o, 32);
  return v;
}
__device__ __forceinline__ float wave_max(float v) {
#pragma unroll
  for (int o = 16; o > 0; o >>= 1) { float t = __shfl_xor(v, o, 32); v = (t > v) ? t : v; }
  return v;
}
__device__ __forceinline__ float gelu_exact(float x) {
  return 0.5f * x * (1.0f + erff(x * 0.70710678118654752f));
}
__device__ __forceinline__ float sigmoidf_(float x) { return 1.0f / (1.0f + expf(-x)); }

// ---------------- utility kernels ----------------
// Wt[c*Kd + k] = bf16(W[k*Nc + c])  (transposed weights for contiguous B fragments)
__global__ __launch_bounds__(256) void cvt_transpose_kernel(const float* __restrict__ W,
                                                            __bf16* __restrict__ Wt,
                                                            int Kd, int Nc) {
  int total = Kd * Nc;
  for (int i = blockIdx.x * 256 + threadIdx.x; i < total; i += gridDim.x * 256) {
    int c = i / Kd, k = i % Kd;
    Wt[i] = bf(W[(size_t)k * Nc + c]);
  }
}

// xb[(b*N+n)*T + t] = x[(b*T+t)*N + n]
__global__ __launch_bounds__(256) void transpose_x_kernel(const float* __restrict__ x,
                                                          __bf16* __restrict__ xb) {
  int total = B_ * N_ * T_;
  for (int i = blockIdx.x * 256 + threadIdx.x; i < total; i += gridDim.x * 256) {
    int b = i / (N_ * T_);
    int r = i % (N_ * T_);
    int n = r / T_, t = r % T_;
    xb[i] = bf(x[((size_t)b * T_ + t) * N_ + n]);
  }
}

// ---------------- generic bf16 WMMA GEMM ----------------
// C[M,Nc] = act(A[M,K] @ W[K,Nc] + bias), W passed TRANSPOSED as Wt[Nc][K].
// Block tile 64(M) x 128(Nc), 8 waves: wave = (rowtile = w&3, colhalf = w>>2),
// 4 accumulators / wave, 4 WMMAs per 32-wide K step.
__global__ __launch_bounds__(256) void gemm_bf16_kernel(
    const __bf16* __restrict__ A, const __bf16* __restrict__ Wt,
    const float* __restrict__ bias, float* __restrict__ Cf, __bf16* __restrict__ Cb,
    int Mrows, int Kdim, int Ncols, int act) {
  __shared__ alignas(16) __bf16 sA[64 * 32];
  __shared__ alignas(16) __bf16 sBt[128 * 40];   // transposed tile, padded stride
  int m0 = blockIdx.x * 64, n0 = blockIdx.y * 128;
  int tid = threadIdx.x, w = tid >> 5;
  int rt = w & 3, ch = w >> 2;
  v8f acc[4] = {{}, {}, {}, {}};
  bool full = (m0 + 64 <= Mrows);
  for (int k0 = 0; k0 < Kdim; k0 += 32) {
#if ASYNC_CP
    if (full) {
      {  // A: 64x32 = 256 chunks of 8 bf16
        int r = tid >> 2, c8 = (tid & 3) * 8;
        async_b128(&A[(size_t)(m0 + r) * Kdim + k0 + c8], &sA[r * 32 + c8]);
      }
#pragma unroll
      for (int j = 0; j < 2; ++j) {  // Bt: 128x32 = 512 chunks
        int ci = tid + j * 256;
        int c = ci >> 2, k8 = (ci & 3) * 8;
        async_b128(&Wt[(size_t)(n0 + c) * Kdim + k0 + k8], &sBt[c * 40 + k8]);
      }
      __builtin_amdgcn_s_wait_asynccnt(0);
    } else
#endif
    {
      for (int i = tid; i < 2048; i += 256) {
        int r = i >> 5, c = i & 31;
        int gm = m0 + r;
        sA[i] = (gm < Mrows) ? A[(size_t)gm * Kdim + k0 + c] : bf(0.0f);
      }
      for (int i = tid; i < 4096; i += 256) {
        int c = i >> 5, kk = i & 31;
        sBt[c * 40 + kk] = Wt[(size_t)(n0 + c) * Kdim + k0 + kk];
      }
    }
    __syncthreads();
    v16bf a = load_A16(sA + rt * 16 * 32, 32);
#pragma unroll
    for (int t = 0; t < 4; ++t) {
      v16bf b = load_Bt16(sBt + (ch * 64 + t * 16) * 40, 40);
      acc[t] = wmma_bf16(a, b, acc[t]);
    }
    __syncthreads();
  }
  int lane = tid & 31, hf = lane >> 4, col = lane & 15;
#pragma unroll
  for (int t = 0; t < 4; ++t) {
    int gc = n0 + ch * 64 + t * 16 + col;
    float bv = bias ? bias[gc] : 0.0f;
#pragma unroll
    for (int r = 0; r < 8; ++r) {
      int gm = m0 + rt * 16 + r + hf * 8;
      if (gm < Mrows) {
        float val = acc[t][r] + bv;
        if (act == 1) val = gelu_exact(val);
        else if (act == 2) val = sigmoidf_(val);
        if (Cf) Cf[(size_t)gm * Ncols + gc] = val;
        if (Cb) Cb[(size_t)gm * Ncols + gc] = bf(val);
      }
    }
  }
}

// ---------------- FFT filter + GELU + LN (in place on h) ----------------
__global__ __launch_bounds__(128) void fft_ln_kernel(float* __restrict__ h,
                                                     const float* __restrict__ freq_w,
                                                     const float* __restrict__ eg,
                                                     const float* __restrict__ eb) {
  int row = blockIdx.x, d = threadIdx.x;
  __shared__ float sX[128], sRe[65], sIm[65], red[4];
  sX[d] = h[(size_t)row * D_ + d];
  __syncthreads();
  if (d < 65) {
    float re = 0.f, im = 0.f;
    for (int t = 0; t < 128; ++t) {
      float s, c;
      sincosf(-TWO_PI_F * (float)(d * t) * (1.0f / 128.0f), &s, &c);
      re += sX[t] * c;
      im += sX[t] * s;
    }
    re *= INV_SQRT128; im *= INV_SQRT128;
    float wr = freq_w[d * 2], wi = freq_w[d * 2 + 1];
    sRe[d] = re * wr - im * wi;
    sIm[d] = re * wi + im * wr;
  }
  __syncthreads();
  float acc = sRe[0];
  for (int f = 1; f < 64; ++f) {
    float s, c;
    sincosf(TWO_PI_F * (float)(f * d) * (1.0f / 128.0f), &s, &c);
    acc += 2.0f * (sRe[f] * c - sIm[f] * s);
  }
  acc += sRe[64] * ((d & 1) ? -1.0f : 1.0f);
  acc *= INV_SQRT128;
  float y = gelu_exact(acc);
  int lane = d & 31, wid = d >> 5;
  float s1 = wave_sum(y);
  if (lane == 0) red[wid] = s1;
  __syncthreads();
  float mu = (red[0] + red[1] + red[2] + red[3]) * (1.0f / 128.0f);
  __syncthreads();
  float diff = y - mu;
  float s2 = wave_sum(diff * diff);
  if (lane == 0) red[wid] = s2;
  __syncthreads();
  float var = (red[0] + red[1] + red[2] + red[3]) * (1.0f / 128.0f);
  h[(size_t)row * D_ + d] = diff * rsqrtf(var + 1e-5f) * eg[d] + eb[d];
}

// ---------------- LayerNorm (+optional (1+scale_p) AdaLN mod) ----------------
__global__ __launch_bounds__(128) void ln_kernel(const float* __restrict__ src,
                                                 const float* __restrict__ g,
                                                 const float* __restrict__ bta,
                                                 const float* __restrict__ pattern,
                                                 int pat_off, float* __restrict__ of,
                                                 __bf16* __restrict__ ob) {
  int row = blockIdx.x, d = threadIdx.x;
  __shared__ float red[4];
  int lane = d & 31, wid = d >> 5;
  float x = src[(size_t)row * D_ + d];
  float s1 = wave_sum(x);
  if (lane == 0) red[wid] = s1;
  __syncthreads();
  float mu = (red[0] + red[1] + red[2] + red[3]) * (1.0f / 128.0f);
  __syncthreads();
  float diff = x - mu;
  float s2 = wave_sum(diff * diff);
  if (lane == 0) red[wid] = s2;
  __syncthreads();
  float var = (red[0] + red[1] + red[2] + red[3]) * (1.0f / 128.0f);
  float y = diff * rsqrtf(var + 1e-5f) * g[d] + bta[d];
  if (pattern) {
    int n = row % N_;
    y *= (1.0f + pattern[(size_t)n * 384 + pat_off + d]);
  }
  if (of) of[(size_t)row * D_ + d] = y;
  if (ob) ob[(size_t)row * D_ + d] = bf(y);
}

// ---------------- per-head softmax (one wave = one head) ----------------
__global__ __launch_bounds__(128) void softmax_head_kernel(float* __restrict__ buf,
                                                           const float* __restrict__ addm) {
  int row = blockIdx.x, tid = threadIdx.x, lane = tid & 31, hh = tid >> 5;
  size_t off = (size_t)row * D_ + hh * DK_ + lane;
  float v = buf[off] * INV_SCALE;
  float mx = wave_max(v);
  float e = expf(v - mx);
  float s = wave_sum(e);
  float r = e / s;
  if (addm) {
    int n = row % N_;
    r += addm[(size_t)n * D_ + hh * DK_ + lane];
  }
  buf[off] = r;
}

// softmax of mem_p (pattern cols 256..383) -> m_sm[N,D]
__global__ __launch_bounds__(128) void m_softmax_kernel(const float* __restrict__ pattern,
                                                        float* __restrict__ m_sm) {
  int n = blockIdx.x, tid = threadIdx.x, lane = tid & 31, hh = tid >> 5;
  float v = pattern[(size_t)n * 384 + 256 + hh * DK_ + lane] * INV_SCALE;
  float mx = wave_max(v);
  float e = expf(v - mx);
  float s = wave_sum(e);
  m_sm[(size_t)n * D_ + hh * DK_ + lane] = e / s;
}

// ---------------- kv = (k+m)^T @ v per (b,h): 32x32, K=N ----------------
__global__ __launch_bounds__(128) void kv_kernel(const float* __restrict__ km,
                                                 const float* __restrict__ v,
                                                 float* __restrict__ kvb) {
  int bh = blockIdx.x, b = bh >> 2, hh = bh & 3;
  __shared__ alignas(16) __bf16 sKT[32 * 32];  // [d][n]  (A: M=d, K=n)
  __shared__ alignas(16) __bf16 sVt[32 * 32];  // [e][n]  (B transposed: col=e, K=n)
  int tid = threadIdx.x, w = tid >> 5;
  int mi = w >> 1, ni = w & 1;
  v8f acc = {};
  for (int n0 = 0; n0 < N_; n0 += 32) {
    for (int i = tid; i < 1024; i += 128) {
      int n = i >> 5, c = i & 31;  // consecutive threads read contiguous d/e
      int gn = n0 + n;
      float kmv = (gn < N_) ? km[((size_t)b * N_ + gn) * D_ + hh * DK_ + c] : 0.0f;
      float vv  = (gn < N_) ? v[((size_t)b * N_ + gn) * D_ + hh * DK_ + c] : 0.0f;
      sKT[c * 32 + n] = bf(kmv);
      sVt[c * 32 + n] = bf(vv);
    }
    __syncthreads();
    v16bf a = load_A16(sKT + (mi * 16) * 32, 32);
    v16bf bb = load_Bt16(sVt + (ni * 16) * 32, 32);
    acc = wmma_bf16(a, bb, acc);
    __syncthreads();
  }
  int lane = tid & 31, hf = lane >> 4, col = lane & 15;
  float* dst = kvb + (size_t)bh * 1024 + (mi * 16) * 32 + ni * 16;
#pragma unroll
  for (int r = 0; r < 8; ++r) dst[(r + hf * 8) * 32 + col] = acc[r];
}

// ---------------- attn = q @ kv per (b,h,ntile): 16x32, K=32 ----------------
__global__ __launch_bounds__(64) void attn_kernel(const float* __restrict__ q,
                                                  const float* __restrict__ kvb,
                                                  __bf16* __restrict__ attnb) {
  int bh = blockIdx.x, b = bh >> 2, hh = bh & 3;
  int n0 = blockIdx.y * 16;
  __shared__ alignas(16) __bf16 sQ[16 * 32];    // [n][d]  (A)
  __shared__ alignas(16) __bf16 sKVt[32 * 32];  // [e][d]  (B transposed)
  int tid = threadIdx.x, w = tid >> 5;
  for (int i = tid; i < 512; i += 64) {
    int r = i >> 5, c = i & 31;
    int gn = n0 + r;
    sQ[i] = (gn < N_) ? bf(q[((size_t)b * N_ + gn) * D_ + hh * DK_ + c]) : bf(0.0f);
  }
  for (int i = tid; i < 1024; i += 64) {
    int e = i >> 5, d = i & 31;
    sKVt[i] = bf(kvb[(size_t)bh * 1024 + d * 32 + e]);
  }
  __syncthreads();
  v16bf a = load_A16(sQ, 32);
  v16bf bb = load_Bt16(sKVt + (w * 16) * 32, 32);
  v8f acc = {};
  acc = wmma_bf16(a, bb, acc);
  int lane = tid & 31, hf = lane >> 4, col = lane & 15;
#pragma unroll
  for (int r = 0; r < 8; ++r) {
    int gn = n0 + r + hf * 8;
    if (gn < N_)
      attnb[((size_t)b * N_ + gn) * D_ + hh * DK_ + w * 16 + col] = bf(acc[r]);
  }
}

// ---------------- elementwise ----------------
__global__ __launch_bounds__(256) void gate_residual_kernel(float* __restrict__ h,
                                                            const float* __restrict__ attn,
                                                            const float* __restrict__ pattern) {
  for (int i = blockIdx.x * 256 + threadIdx.x; i < BND_; i += gridDim.x * 256) {
    int d = i & 127, n = (i >> 7) % N_;
    h[i] += sigmoidf_(pattern[(size_t)n * 384 + d]) * attn[i];
  }
}
__global__ __launch_bounds__(256) void add_residual_kernel(float* __restrict__ h,
                                                           const float* __restrict__ f) {
  for (int i = blockIdx.x * 256 + threadIdx.x; i < BND_; i += gridDim.x * 256) h[i] += f[i];
}
__global__ __launch_bounds__(256) void cat_kernel(const float* __restrict__ h,
                                                  const float* __restrict__ epi,
                                                  __bf16* __restrict__ catb) {
  int total = BN_ * 256;
  for (int i = blockIdx.x * 256 + threadIdx.x; i < total; i += gridDim.x * 256) {
    int row = i >> 8, c = i & 255;
    float v = (c < 128) ? h[(size_t)row * D_ + c] : epi[(size_t)row * D_ + (c - 128)];
    catb[i] = bf(v);
  }
}
__global__ __launch_bounds__(256) void fuse_kernel(const float* __restrict__ g,
                                                   const float* __restrict__ epi,
                                                   const float* __restrict__ h,
                                                   __bf16* __restrict__ fusedb) {
  for (int i = blockIdx.x * 256 + threadIdx.x; i < BND_; i += gridDim.x * 256) {
    float gv = g[i];
    fusedb[i] = bf(gv * epi[i] + (1.0f - gv) * h[i]);
  }
}

// ---------------- pooled mean + L2-normalize ----------------
__global__ __launch_bounds__(128) void pool_norm_kernel(const float* __restrict__ src,
                                                        float* __restrict__ dst, int count) {
  int row = blockIdx.x, d = threadIdx.x;
  __shared__ float red[4];
  const float* p = src + (size_t)row * count * D_ + d;
  float s = 0.f;
  for (int n = 0; n < count; ++n) s += p[(size_t)n * D_];
  s *= (1.0f / (float)count);
  int lane = d & 31, wid = d >> 5;
  float sq = wave_sum(s * s);
  if (lane == 0) red[wid] = sq;
  __syncthreads();
  float nrm = sqrtf(red[0] + red[1] + red[2] + red[3]);
  dst[(size_t)row * D_ + d] = s / (nrm + 1e-8f);
}

__global__ __launch_bounds__(256) void sims_kernel(const float* __restrict__ qn,
                                                   const float* __restrict__ mn,
                                                   float* __restrict__ sims) {
  int b = blockIdx.x, tid = threadIdx.x;
  __shared__ float sq[128];
  if (tid < 128) sq[tid] = qn[(size_t)b * D_ + tid];
  __syncthreads();
  for (int m = tid; m < M_; m += 256) {
    float dot = 0.f;
    const float* mp = mn + (size_t)m * D_;
#pragma unroll 8
    for (int d = 0; d < 128; ++d) dot += sq[d] * mp[d];
    sims[(size_t)b * M_ + m] = dot;
  }
}

__global__ __launch_bounds__(256) void topk_kernel(float* __restrict__ sims,
                                                   int* __restrict__ idxout) {
  int b = blockIdx.x, tid = threadIdx.x;
  __shared__ float sv[M_];
  __shared__ float bval[256];
  __shared__ int bidx[256];
  for (int i = tid; i < M_; i += 256) sv[i] = sims[(size_t)b * M_ + i];
  __syncthreads();
  for (int kk = 0; kk < K_; ++kk) {
    float best = -1e30f;
    int bi = M_;
    for (int i = tid; i < M_; i += 256) {
      float v = sv[i];
      if (v > best || (v == best && i < bi)) { best = v; bi = i; }
    }
    bval[tid] = best; bidx[tid] = bi;
    __syncthreads();
    for (int s = 128; s > 0; s >>= 1) {
      if (tid < s) {
        if (bval[tid + s] > bval[tid] ||
            (bval[tid + s] == bval[tid] && bidx[tid + s] < bidx[tid])) {
          bval[tid] = bval[tid + s];
          bidx[tid] = bidx[tid + s];
        }
      }
      __syncthreads();
    }
    if (tid == 0) {
      idxout[b * K_ + kk] = bidx[0];
      sv[bidx[0]] = -1e30f;
    }
    __syncthreads();
  }
}

// ---------------- fused episodic MHA (gather + 3 WMMA GEMMs + softmax) ----------------
// grid (B, NT), block 256, dynamic LDS 176160 B (CDNA5 320KB WGP LDS).
// Weights passed TRANSPOSED (Wt[col][k], 128x128 bf16).
__global__ __launch_bounds__(256) void episodic_kernel(
    const float* __restrict__ memory_bank, const int* __restrict__ idxb,
    const float* __restrict__ h, const __bf16* __restrict__ ewq_t,
    const __bf16* __restrict__ ewk_t, const __bf16* __restrict__ ewv_t,
    const float* __restrict__ ebq, const float* __restrict__ ebk,
    const float* __restrict__ ebv, __bf16* __restrict__ eheadsb) {
  extern __shared__ char smem[];
  __bf16* sA  = (__bf16*)smem;                // 128x128 bf16 : gathered kvm rows
  __bf16* sH  = (__bf16*)(smem + 32768);      // 16x128 bf16  : query rows
  float*  sKH = (float*)(smem + 36864);       // 128x128 f32
  float*  sVH = (float*)(smem + 102400);      // 128x128 f32
  float*  sQH = (float*)(smem + 167936);      // 16x128 f32
  int*    sIdx = (int*)(smem + 176128);       // 8 ints

  int b = blockIdx.x, n0 = blockIdx.y * 16;
  int tid = threadIdx.x, w = tid >> 5;
  if (tid < K_) sIdx[tid] = idxb[b * K_ + tid];
  __syncthreads();

  // gather A rows: row r = n*8 + k  ->  memory_bank[idx[b,k], n0+n, :]
  for (int i = tid; i < 16384; i += 256) {
    int r = i >> 7, c = i & 127;
    int n = r >> 3, k = r & 7;
    int gn = n0 + n; if (gn >= N_) gn = N_ - 1;
    sA[i] = bf(memory_bank[((size_t)sIdx[k] * N_ + gn) * D_ + c]);
  }
  for (int i = tid; i < 2048; i += 256) {
    int r = i >> 7, c = i & 127;
    int gn = n0 + r; if (gn >= N_) gn = N_ - 1;
    sH[i] = bf(h[((size_t)b * N_ + gn) * D_ + c]);
  }
  __syncthreads();

  // KH = A @ eWk + ebk ; VH = A @ eWv + ebv  (wave = 16-row stripe)
#pragma unroll 1
  for (int ct = 0; ct < 8; ++ct) {
    v8f accK = {}, accV = {};
#pragma unroll
    for (int kt = 0; kt < 4; ++kt) {
      v16bf a = load_A16(sA + (w * 16) * 128 + kt * 32, 128);
      v16bf bk_ = load_Bt16(ewk_t + (size_t)(ct * 16) * 128 + kt * 32, 128);
      v16bf bv_ = load_Bt16(ewv_t + (size_t)(ct * 16) * 128 + kt * 32, 128);
      accK = wmma_bf16(a, bk_, accK);
      accV = wmma_bf16(a, bv_, accV);
    }
    store_c_bias_lds(sKH + (w * 16) * 128 + ct * 16, 128, accK, ebk, ct * 16);
    store_c_bias_lds(sVH + (w * 16) * 128 + ct * 16, 128, accV, ebv, ct * 16);
  }
  // QH = Hrows @ eWq + ebq  (wave w -> col tile w)
  {
    v8f accQ = {};
#pragma unroll
    for (int kt = 0; kt < 4; ++kt) {
      v16bf a = load_A16(sH + kt * 32, 128);
      v16bf bq_ = load_Bt16(ewq_t + (size_t)(w * 16) * 128 + kt * 32, 128);
      accQ = wmma_bf16(a, bq_, accQ);
    }
    store_c_bias_lds(sQH + w * 16, 128, accQ, ebq, w * 16);
  }
  __syncthreads();

  // per (n, head) softmax attention over K=8 neighbors
  if (tid < 64) {
    int n = tid >> 2, hh = tid & 3;
    float sc[K_];
    float mx = -1e30f;
#pragma unroll
    for (int k = 0; k < K_; ++k) {
      int r = n * 8 + k;
      float s = 0.f;
#pragma unroll 8
      for (int d = 0; d < DK_; ++d)
        s += sQH[n * 128 + hh * DK_ + d] * sKH[r * 128 + hh * DK_ + d];
      s *= INV_SCALE;
      sc[k] = s;
      mx = (s > mx) ? s : mx;
    }
    float sum = 0.f;
#pragma unroll
    for (int k = 0; k < K_; ++k) { sc[k] = expf(sc[k] - mx); sum += sc[k]; }
    float inv = 1.0f / sum;
    int gn = n0 + n;
    if (gn < N_) {
      for (int d = 0; d < DK_; ++d) {
        float o = 0.f;
#pragma unroll
        for (int k = 0; k < K_; ++k)
          o += sc[k] * sVH[(n * 8 + k) * 128 + hh * DK_ + d];
        eheadsb[((size_t)b * N_ + gn) * D_ + hh * DK_ + d] = bf(o * inv);
      }
    }
  }
}

// ---------------- host ----------------
extern "C" void kernel_launch(void* const* d_in, const int* in_sizes, int n_in,
                              void* d_out, int out_size, void* d_ws, size_t ws_size,
                              hipStream_t stream) {
  const float* x        = (const float*)d_in[0];
  const float* W_fconv  = (const float*)d_in[1];
  const float* b_fconv  = (const float*)d_in[2];
  const float* freq_w   = (const float*)d_in[3];
  const float* enc_g    = (const float*)d_in[4];
  const float* enc_b    = (const float*)d_in[5];
  const float* pattern  = (const float*)d_in[6];
  const float* ln1_g    = (const float*)d_in[7];
  const float* ln1_b    = (const float*)d_in[8];
  const float* ln2_g    = (const float*)d_in[9];
  const float* ln2_b    = (const float*)d_in[10];
  const float* Wq = (const float*)d_in[11]; const float* bq = (const float*)d_in[12];
  const float* Wk = (const float*)d_in[13]; const float* bk = (const float*)d_in[14];
  const float* Wv = (const float*)d_in[15]; const float* bv = (const float*)d_in[16];
  const float* Wo = (const float*)d_in[17]; const float* bo = (const float*)d_in[18];
  const float* W1 = (const float*)d_in[19]; const float* b1 = (const float*)d_in[20];
  const float* W2 = (const float*)d_in[21]; const float* b2 = (const float*)d_in[22];
  const float* eWq = (const float*)d_in[23]; const float* ebq = (const float*)d_in[24];
  const float* eWk = (const float*)d_in[25]; const float* ebk = (const float*)d_in[26];
  const float* eWv = (const float*)d_in[27]; const float* ebv = (const float*)d_in[28];
  const float* eWo = (const float*)d_in[29]; const float* ebo = (const float*)d_in[30];
  const float* Wg = (const float*)d_in[31]; const float* bg = (const float*)d_in[32];
  const float* Wp = (const float*)d_in[33]; const float* bp = (const float*)d_in[34];
  const float* memory_bank = (const float*)d_in[35];
  float* out = (float*)d_out;

  // ---- workspace carve ----
  char* p = (char*)d_ws;
  auto alloc = [&](size_t bytes) -> void* {
    void* r = (void*)p;
    p += (bytes + 255) & ~(size_t)255;
    return r;
  };
  __bf16* xb     = (__bf16*)alloc((size_t)B_ * N_ * T_ * 2);
  float*  h      = (float*)alloc((size_t)BND_ * 4);
  __bf16* h1b    = (__bf16*)alloc((size_t)BND_ * 2);
  float*  qlin   = (float*)alloc((size_t)BND_ * 4);
  float*  klin   = (float*)alloc((size_t)BND_ * 4);
  float*  vlin   = (float*)alloc((size_t)BND_ * 4);
  float*  kvb    = (float*)alloc((size_t)B_ * H_ * DK_ * DK_ * 4);
  __bf16* attnb  = (__bf16*)alloc((size_t)BND_ * 2);
  float*  tmpf   = (float*)alloc((size_t)BND_ * 4);   // attn_out / ffn_out
  __bf16* h2b    = (__bf16*)alloc((size_t)BND_ * 2);
  __bf16* gmidb  = (__bf16*)alloc((size_t)BN_ * D4_ * 2);
  float*  m_sm   = (float*)alloc((size_t)N_ * D_ * 4);
  float*  qn     = (float*)alloc((size_t)B_ * D_ * 4);
  float*  mn     = (float*)alloc((size_t)M_ * D_ * 4);
  float*  sims   = (float*)alloc((size_t)B_ * M_ * 4);
  int*    idxb   = (int*)alloc((size_t)B_ * K_ * 4);
  __bf16* ehb    = (__bf16*)alloc((size_t)BND_ * 2);
  float*  epi    = (float*)alloc((size_t)BND_ * 4);
  __bf16* catb   = (__bf16*)alloc((size_t)BN_ * 256 * 2);
  float*  gbuf   = (float*)alloc((size_t)BND_ * 4);
  __bf16* fusedb = (__bf16*)alloc((size_t)BND_ * 2);
  // transposed bf16 weights Wt[Nc][K]
  __bf16* wfc_t = (__bf16*)alloc((size_t)D_ * T_ * 2);
  __bf16* wq_t  = (__bf16*)alloc((size_t)D_ * D_ * 2);
  __bf16* wk_t  = (__bf16*)alloc((size_t)D_ * D_ * 2);
  __bf16* wv_t  = (__bf16*)alloc((size_t)D_ * D_ * 2);
  __bf16* wo_t  = (__bf16*)alloc((size_t)D_ * D_ * 2);
  __bf16* w1_t  = (__bf16*)alloc((size_t)D4_ * D_ * 2);
  __bf16* w2_t  = (__bf16*)alloc((size_t)D_ * D4_ * 2);
  __bf16* ewq_t = (__bf16*)alloc((size_t)D_ * D_ * 2);
  __bf16* ewk_t = (__bf16*)alloc((size_t)D_ * D_ * 2);
  __bf16* ewv_t = (__bf16*)alloc((size_t)D_ * D_ * 2);
  __bf16* ewo_t = (__bf16*)alloc((size_t)D_ * D_ * 2);
  __bf16* wg_t  = (__bf16*)alloc((size_t)D_ * 256 * 2);
  __bf16* wp_t  = (__bf16*)alloc((size_t)D_ * D_ * 2);
  (void)in_sizes; (void)n_in; (void)out_size; (void)ws_size;

  auto cvtT = [&](const float* s, __bf16* dst, int Kd, int Nc) {
    int n = Kd * Nc;
    cvt_transpose_kernel<<<(n + 255) / 256, 256, 0, stream>>>(s, dst, Kd, Nc);
  };
  cvtT(W_fconv, wfc_t, T_, D_);
  cvtT(Wq, wq_t, D_, D_);   cvtT(Wk, wk_t, D_, D_);
  cvtT(Wv, wv_t, D_, D_);   cvtT(Wo, wo_t, D_, D_);
  cvtT(W1, w1_t, D_, D4_);  cvtT(W2, w2_t, D4_, D_);
  cvtT(eWq, ewq_t, D_, D_); cvtT(eWk, ewk_t, D_, D_);
  cvtT(eWv, ewv_t, D_, D_); cvtT(eWo, ewo_t, D_, D_);
  cvtT(Wg, wg_t, 256, D_);  cvtT(Wp, wp_t, D_, D_);

  auto gemm = [&](const __bf16* A, const __bf16* Wt, const float* bias, float* Cf,
                  __bf16* Cb, int Mr, int Kd, int Nc, int act) {
    dim3 grid((Mr + 63) / 64, Nc / 128);
    gemm_bf16_kernel<<<grid, 256, 0, stream>>>(A, Wt, bias, Cf, Cb, Mr, Kd, Nc, act);
  };

  // 1) temporal encoder GEMM + FFT filter + gelu + LN
  transpose_x_kernel<<<4096, 256, 0, stream>>>(x, xb);
  gemm(xb, wfc_t, b_fconv, h, nullptr, BN_, T_, D_, 0);
  fft_ln_kernel<<<BN_, 128, 0, stream>>>(h, freq_w, enc_g, enc_b);

  // 2) AdaLN + dual-path linear attention
  m_softmax_kernel<<<N_, 128, 0, stream>>>(pattern, m_sm);
  ln_kernel<<<BN_, 128, 0, stream>>>(h, ln1_g, ln1_b, pattern, 128, nullptr, h1b);
  gemm(h1b, wq_t, bq, qlin, nullptr, BN_, D_, D_, 0);
  gemm(h1b, wk_t, bk, klin, nullptr, BN_, D_, D_, 0);
  gemm(h1b, wv_t, bv, vlin, nullptr, BN_, D_, D_, 0);
  softmax_head_kernel<<<BN_, 128, 0, stream>>>(qlin, nullptr);
  softmax_head_kernel<<<BN_, 128, 0, stream>>>(klin, m_sm);  // klin -> k+m
  kv_kernel<<<B_ * H_, 128, 0, stream>>>(klin, vlin, kvb);
  attn_kernel<<<dim3(B_ * H_, NT_), 64, 0, stream>>>(qlin, kvb, attnb);
  gemm(attnb, wo_t, bo, tmpf, nullptr, BN_, D_, D_, 0);
  gate_residual_kernel<<<10400, 256, 0, stream>>>(h, tmpf, pattern);

  // 3) FFN residual
  ln_kernel<<<BN_, 128, 0, stream>>>(h, ln2_g, ln2_b, nullptr, 0, nullptr, h2b);
  gemm(h2b, w1_t, b1, nullptr, gmidb, BN_, D_, D4_, 1);      // gelu
  gemm(gmidb, w2_t, b2, tmpf, nullptr, BN_, D4_, D_, 0);
  add_residual_kernel<<<10400, 256, 0, stream>>>(h, tmpf);

  // 4) retrieval: pooled means, cosine sims, top-K
  pool_norm_kernel<<<B_, 128, 0, stream>>>(h, qn, N_);
  pool_norm_kernel<<<M_, 128, 0, stream>>>(memory_bank, mn, N_);
  sims_kernel<<<B_, 256, 0, stream>>>(qn, mn, sims);
  topk_kernel<<<B_, 256, 0, stream>>>(sims, idxb);

  // 5) fused episodic MHA (gather + WMMA in 176KB LDS)
  episodic_kernel<<<dim3(B_, NT_), 256, 176160, stream>>>(
      memory_bank, idxb, h, ewq_t, ewk_t, ewv_t, ebq, ebk, ebv, ehb);
  gemm(ehb, ewo_t, ebo, epi, nullptr, BN_, D_, D_, 0);

  // 6) gated fusion + final projection
  cat_kernel<<<20800, 256, 0, stream>>>(h, epi, catb);
  gemm(catb, wg_t, bg, gbuf, nullptr, BN_, 256, D_, 2);       // sigmoid
  fuse_kernel<<<10400, 256, 0, stream>>>(gbuf, epi, h, fusedb);
  gemm(fusedb, wp_t, bp, out, nullptr, BN_, D_, D_, 0);
}
